// LatentODEBlock_53601191854787
// MI455X (gfx1250) — compile-verified, matrix operand order checked by reading, not run on hardware
//
#include <hip/hip_runtime.h>
#include <stdint.h>

// LatentODE scan: B=1024 rows, N=512 steps, MLP 160->128->128->128->64 with LN+tanh.
// One block = 16 batch rows for all 511 steps. Weights live in registers as WMMA
// B-fragments (bf16); state/activations live in LDS; fp32 accumulation via
// v_wmma_f32_16x16x32_bf16. Per-step condition slices are double-buffered in LDS
// via GLOBAL_LOAD_ASYNC_TO_LDS_B128 (ASYNCcnt), overlapping the per-step cond
// global read with the whole step's compute.

#define LATENT 64
#define CONDD  96
#define NEUR   128
#define DIN0   (LATENT + CONDD)   // 160
#define NSTEP  512
#define BROWS  16
// padded LDS strides (bytes per row divisible by 16 for b128 loads, and
// bank-skewed so 16 rows don't collide)
#define XS 168   // xbuf stride in bf16 elems (336 B/row)
#define HS 136   // hbuf stride in bf16 elems (272 B/row)
#define PS 132   // hpre stride in f32 elems
#define ZS 68    // z/dz stride in f32 elems
#define CSTG 128 // cond staging stride in f32 elems (512 B/row, 96 used + pad)

typedef __bf16 bf16_t;
typedef __attribute__((ext_vector_type(16))) __bf16 v16bf;
typedef __attribute__((ext_vector_type(8)))  __bf16 v8bf;
typedef __attribute__((ext_vector_type(8)))  float  v8f;

__device__ __forceinline__ bf16_t f2bf(float f) {
  unsigned u = __builtin_bit_cast(unsigned, f);
  unsigned r = u + 0x7FFFu + ((u >> 16) & 1u);   // round-to-nearest-even
  unsigned short h = (unsigned short)(r >> 16);
  return __builtin_bit_cast(bf16_t, h);
}

__device__ __forceinline__ float tanh_fast(float x) {
  float cx = fminf(fmaxf(x, -9.f), 9.f);
  float e = __expf(2.f * cx);
  return (e - 1.f) / (e + 1.f);
}

__device__ __forceinline__ v8f zero8() {
  v8f z;
  #pragma unroll
  for (int i = 0; i < 8; ++i) z[i] = 0.f;
  return z;
}

// async global -> LDS 16B chunk; tracked with ASYNCcnt (CDNA5 ISA).
// vdst = LDS byte address, vaddr = 32-bit byte offset, saddr = 64-bit base (GVS).
__device__ __forceinline__ void async_ld_b128(uint64_t gbase, unsigned ldsoff,
                                              unsigned goff) {
  asm volatile("global_load_async_to_lds_b128 %0, %1, %2"
               :: "v"(ldsoff), "v"(goff), "s"(gbase) : "memory");
}

__device__ __forceinline__ void wait_asynccnt_le2() {
  asm volatile("s_wait_asynccnt 0x2" ::: "memory");
}

// B-fragment (K x 16 tile of weight matrix, bf16) in ISA layout:
// lane holds column `col`; lanes 0-15 carry K = kbase+{0..7,16..23},
// lanes 16-31 carry K = kbase+{8..15,24..31}.
__device__ __forceinline__ v16bf load_bfrag(const float* __restrict__ W, int dout,
                                            int kbase, int col, int lhi) {
  v16bf r;
  int k0 = kbase + (lhi ? 8 : 0);
  #pragma unroll
  for (int j = 0; j < 8; ++j) r[j]     = f2bf(W[(k0 + j)      * dout + col]);
  #pragma unroll
  for (int j = 0; j < 8; ++j) r[8 + j] = f2bf(W[(k0 + 16 + j) * dout + col]);
  return r;
}

// A-fragment (16 x 32 tile of activations) from row-major bf16 LDS buffer:
// lane holds row = lane&15; two aligned 16B chunks per K-tile.
__device__ __forceinline__ v16bf load_afrag(const bf16_t* __restrict__ buf, int stride,
                                            int kbase, int row, int lhi) {
  int k0 = kbase + (lhi ? 8 : 0);
  const v8bf* p = (const v8bf*)(buf + row * stride + k0);
  v8bf a0 = p[0];
  v8bf a1 = p[2];  // +16 bf16 elements
  return __builtin_shufflevector(a0, a1, 0,1,2,3,4,5,6,7,8,9,10,11,12,13,14,15);
}

// fused bias + layernorm + tanh; 16 threads per row, 8 cols per thread
__device__ __forceinline__ void ln_tanh_pass(const float* hpre, bf16_t* hbuf,
                                             const float* lnp, int layer, int tid) {
  int row = tid >> 4;
  int cg  = tid & 15;
  const float* bb = lnp + layer * 384;  // [bias | gamma | beta], 128 each
  float vals[8];
  float s = 0.f, s2 = 0.f;
  #pragma unroll
  for (int j = 0; j < 8; ++j) {
    int c = cg * 8 + j;
    float v = hpre[row * PS + c] + bb[c];
    vals[j] = v; s += v; s2 += v * v;
  }
  #pragma unroll
  for (int off = 1; off < 16; off <<= 1) {
    s  += __shfl_xor(s,  off, 16);
    s2 += __shfl_xor(s2, off, 16);
  }
  float mean = s * (1.f / 128.f);
  float var  = s2 * (1.f / 128.f) - mean * mean;
  float rstd = rsqrtf(var + 1e-5f);
  #pragma unroll
  for (int j = 0; j < 8; ++j) {
    int c = cg * 8 + j;
    float y = (vals[j] - mean) * rstd * bb[128 + c] + bb[256 + c];
    hbuf[row * HS + c] = f2bf(tanh_fast(y));
  }
}

__global__ __launch_bounds__(256)
void latent_ode_kernel(const float* __restrict__ z0,
                       const float* __restrict__ tgrid,
                       const float* __restrict__ cond,
                       const float* __restrict__ W0, const float* __restrict__ b0v,
                       const float* __restrict__ g0v, const float* __restrict__ be0v,
                       const float* __restrict__ W1, const float* __restrict__ b1v,
                       const float* __restrict__ g1v, const float* __restrict__ be1v,
                       const float* __restrict__ W2, const float* __restrict__ b2v,
                       const float* __restrict__ g2v, const float* __restrict__ be2v,
                       const float* __restrict__ Wo, const float* __restrict__ bov,
                       float* __restrict__ out) {
  __shared__ __align__(16) bf16_t xbuf[BROWS * XS];        // [z | cond] bf16
  __shared__ __align__(16) bf16_t hbuf[BROWS * HS];        // post-LN activations bf16
  __shared__ __align__(16) float  hpre[BROWS * PS];        // pre-LN WMMA output f32
  __shared__ __align__(16) float  condstg[2 * BROWS * CSTG]; // cond double buffer f32
  __shared__ float zbuf[BROWS * ZS];                       // recurrent state f32
  __shared__ float dzbuf[BROWS * ZS];
  __shared__ float lnp[3 * 384];                           // per-layer bias/gamma/beta
  __shared__ float bo_s[LATENT];

  const int tid  = threadIdx.x;
  const int wave = tid >> 5;
  const int lane = tid & 31;
  const int lhi  = lane >> 4;
  const int lcol = lane & 15;
  const int b0r  = blockIdx.x * BROWS;
  const uint64_t condp = (uint64_t)(uintptr_t)cond;

  // --- one-time preload: LN params ---
  if (tid < 128) {
    lnp[0 * 384 +   0 + tid] = b0v[tid];
    lnp[0 * 384 + 128 + tid] = g0v[tid];
    lnp[0 * 384 + 256 + tid] = be0v[tid];
    lnp[1 * 384 +   0 + tid] = b1v[tid];
    lnp[1 * 384 + 128 + tid] = g1v[tid];
    lnp[1 * 384 + 256 + tid] = be1v[tid];
    lnp[2 * 384 +   0 + tid] = b2v[tid];
    lnp[2 * 384 + 128 + tid] = g2v[tid];
    lnp[2 * 384 + 256 + tid] = be2v[tid];
  }
  if (tid < LATENT) bo_s[tid] = bov[tid];

  // --- one-time: z0 into state + trajectory[:,0,:] ---
  for (int e = tid; e < BROWS * LATENT; e += 256) {
    int r = e >> 6, c = e & 63;
    float v = z0[(size_t)(b0r + r) * LATENT + c];
    zbuf[r * ZS + c] = v;
    out[((size_t)(b0r + r) * NSTEP + 0) * LATENT + c] = v;
  }

  // --- prologue: async-stage cond slice for step 0 into buffer 0 ---
  // 512 x 16B chunks over 256 threads: every wave issues exactly 2 async loads.
  #pragma unroll
  for (int c2 = 0; c2 < 2; ++c2) {
    int ch  = tid + c2 * 256;
    int row = ch >> 5, q = ch & 31;            // q: 16B chunk within padded row
    unsigned goff = (unsigned)((((size_t)(b0r + row)) * NSTEP + 0) * (CONDD * 4))
                  + (unsigned)((q < 24 ? q : 0) * 16);   // clamp padding chunks
    unsigned ldsoff = (unsigned)(uintptr_t)&condstg[row * CSTG + q * 4];
    async_ld_b128(condp, ldsoff, goff);
  }

  // --- one-time: weights -> register-resident bf16 B-fragments ---
  v16bf w0f[5], w1f[4], w2f[4], wof[4];
  {
    int col = wave * 16 + lcol;          // 0..127
    #pragma unroll
    for (int kt = 0; kt < 5; ++kt) w0f[kt] = load_bfrag(W0, NEUR, kt * 32, col, lhi);
    #pragma unroll
    for (int kt = 0; kt < 4; ++kt) w1f[kt] = load_bfrag(W1, NEUR, kt * 32, col, lhi);
    #pragma unroll
    for (int kt = 0; kt < 4; ++kt) w2f[kt] = load_bfrag(W2, NEUR, kt * 32, col, lhi);
    int ocol = (wave & 3) * 16 + lcol;   // 0..63 (Wo used by waves 0-3)
    #pragma unroll
    for (int kt = 0; kt < 4; ++kt) wof[kt] = load_bfrag(Wo, LATENT, kt * 32, ocol, lhi);
  }
  __syncthreads();

  // --- sequential scan over 511 Euler steps ---
  #pragma unroll 1
  for (int i = 0; i < NSTEP - 1; ++i) {
    const int cb = i & 1;                 // cond buffer to consume this step

    // Issue async staging of next step's cond into the other buffer (always 2
    // issues per wave, so "asynccnt <= 2" below exactly drains the previous
    // step's loads while the new ones overlap this step's compute).
    {
      int pstep = (i + 1 < NSTEP - 1) ? (i + 1) : (NSTEP - 2);
      int pb    = (i + 1) & 1;
      #pragma unroll
      for (int c2 = 0; c2 < 2; ++c2) {
        int ch  = tid + c2 * 256;
        int row = ch >> 5, q = ch & 31;
        unsigned goff = (unsigned)((((size_t)(b0r + row)) * NSTEP + pstep) * (CONDD * 4))
                      + (unsigned)((q < 24 ? q : 0) * 16);
        unsigned ldsoff = (unsigned)(uintptr_t)&condstg[pb * (BROWS * CSTG) + row * CSTG + q * 4];
        async_ld_b128(condp, ldsoff, goff);
      }
      wait_asynccnt_le2();                // step-i cond chunks now in LDS
    }
    __syncthreads();                      // make all waves' async writes visible

    // Stage A: xbuf = bf16([z | cond_i]) from LDS
    for (int e = tid; e < BROWS * DIN0; e += 256) {
      int r = e / DIN0, c = e - r * DIN0;
      float v = (c < LATENT)
                    ? zbuf[r * ZS + c]
                    : condstg[cb * (BROWS * CSTG) + r * CSTG + (c - LATENT)];
      xbuf[r * XS + c] = f2bf(v);
    }
    __syncthreads();

    // GEMM 0: [16x160] @ [160x128]
    {
      v8f acc = zero8();
      #pragma unroll
      for (int kt = 0; kt < 5; ++kt) {
        v16bf a = load_afrag(xbuf, XS, kt * 32, lcol, lhi);
        acc = __builtin_amdgcn_wmma_f32_16x16x32_bf16(false, a, false, w0f[kt],
                                                      (short)0, acc, false, false);
      }
      #pragma unroll
      for (int r = 0; r < 8; ++r)
        hpre[(r + 8 * lhi) * PS + wave * 16 + lcol] = acc[r];
    }
    __syncthreads();
    ln_tanh_pass(hpre, hbuf, lnp, 0, tid);
    __syncthreads();

    // GEMM 1: [16x128] @ [128x128]
    {
      v8f acc = zero8();
      #pragma unroll
      for (int kt = 0; kt < 4; ++kt) {
        v16bf a = load_afrag(hbuf, HS, kt * 32, lcol, lhi);
        acc = __builtin_amdgcn_wmma_f32_16x16x32_bf16(false, a, false, w1f[kt],
                                                      (short)0, acc, false, false);
      }
      #pragma unroll
      for (int r = 0; r < 8; ++r)
        hpre[(r + 8 * lhi) * PS + wave * 16 + lcol] = acc[r];
    }
    __syncthreads();
    ln_tanh_pass(hpre, hbuf, lnp, 1, tid);
    __syncthreads();

    // GEMM 2: [16x128] @ [128x128]
    {
      v8f acc = zero8();
      #pragma unroll
      for (int kt = 0; kt < 4; ++kt) {
        v16bf a = load_afrag(hbuf, HS, kt * 32, lcol, lhi);
        acc = __builtin_amdgcn_wmma_f32_16x16x32_bf16(false, a, false, w2f[kt],
                                                      (short)0, acc, false, false);
      }
      #pragma unroll
      for (int r = 0; r < 8; ++r)
        hpre[(r + 8 * lhi) * PS + wave * 16 + lcol] = acc[r];
    }
    __syncthreads();
    ln_tanh_pass(hpre, hbuf, lnp, 2, tid);
    __syncthreads();

    // Output head: [16x128] @ [128x64] (waves 0-3; wave-uniform branch keeps EXEC full)
    if (wave < 4) {
      v8f acc = zero8();
      #pragma unroll
      for (int kt = 0; kt < 4; ++kt) {
        v16bf a = load_afrag(hbuf, HS, kt * 32, lcol, lhi);
        acc = __builtin_amdgcn_wmma_f32_16x16x32_bf16(false, a, false, wof[kt],
                                                      (short)0, acc, false, false);
      }
      #pragma unroll
      for (int r = 0; r < 8; ++r)
        dzbuf[(r + 8 * lhi) * ZS + wave * 16 + lcol] = acc[r];
    }
    __syncthreads();

    // Stage G: Euler update + trajectory write
    for (int e = tid; e < BROWS * LATENT; e += 256) {
      int r = e >> 6, c = e & 63;
      int gb = b0r + r;
      float dt = tgrid[(size_t)gb * NSTEP + (i + 1)] - tgrid[(size_t)gb * NSTEP + i];
      float dz = dzbuf[r * ZS + c] + bo_s[c];
      float zn = zbuf[r * ZS + c] + dt * dz;
      zbuf[r * ZS + c] = zn;
      out[((size_t)gb * NSTEP + (i + 1)) * LATENT + c] = zn;
    }
    __syncthreads();
  }
}

extern "C" void kernel_launch(void* const* d_in, const int* in_sizes, int n_in,
                              void* d_out, int out_size, void* d_ws, size_t ws_size,
                              hipStream_t stream) {
  (void)n_in; (void)out_size; (void)d_ws; (void)ws_size;
  const float* z0   = (const float*)d_in[0];
  const float* t    = (const float*)d_in[1];
  const float* cond = (const float*)d_in[2];
  const float* W0   = (const float*)d_in[3];
  const float* b0   = (const float*)d_in[4];
  const float* g0   = (const float*)d_in[5];
  const float* be0  = (const float*)d_in[6];
  const float* W1   = (const float*)d_in[7];
  const float* b1   = (const float*)d_in[8];
  const float* g1   = (const float*)d_in[9];
  const float* be1  = (const float*)d_in[10];
  const float* W2   = (const float*)d_in[11];
  const float* b2   = (const float*)d_in[12];
  const float* g2   = (const float*)d_in[13];
  const float* be2  = (const float*)d_in[14];
  const float* Wo   = (const float*)d_in[15];
  const float* bo   = (const float*)d_in[16];

  int B = in_sizes[0] / LATENT;  // 1024
  dim3 grid(B / BROWS);          // 64 blocks of 16 batch rows
  latent_ode_kernel<<<grid, dim3(256), 0, stream>>>(
      z0, t, cond, W0, b0, g0, be0, W1, b1, g1, be1, W2, b2, g2, be2,
      Wo, bo, (float*)d_out);
}
